// CopyGenerator_55783035240973
// MI455X (gfx1250) — compile-verified, matrix operand order checked by reading, not run on hardware
//
#include <hip/hip_runtime.h>
#include <hip/hip_bf16.h>
#include <stdint.h>

typedef __attribute__((ext_vector_type(16))) __bf16 v16bf;
typedef __attribute__((ext_vector_type(8)))  float  v8f;

#define BM 128
#define BN 128
#define BK 32
#define APITCH 40   // ushorts per LDS row: 32 data + 8 pad (80B, 16B-aligned)

// Pack two f32 -> two bf16 (round-half-up; bias << bf16 quantization noise).
// One v_add per input + one v_perm_b32 for the pack.
__device__ __forceinline__ unsigned pack2_bf16(float x, float y) {
    unsigned a = __float_as_uint(x) + 0x8000u;   // low  half of result
    unsigned b = __float_as_uint(y) + 0x8000u;   // high half of result
#if __has_builtin(__builtin_amdgcn_perm)
    // result bytes {b3,b2,a3,a2}: sel 0-3 -> src1(a), 4-7 -> src0(b)
    return __builtin_amdgcn_perm(b, a, 0x07060302u);
#else
    return (a >> 16) | (b & 0xFFFF0000u);
#endif
}

__device__ __forceinline__ unsigned short bf16_hi(float x) {
    return (unsigned short)((__float_as_uint(x) + 0x8000u) >> 16); // -> ds_store_b16_d16_hi
}

// ---------------- GEMM: logits = hidden @ W + b  (bf16 WMMA, f32 accum) ---------------
__global__ __launch_bounds__(256)
void gemm_logits(const float* __restrict__ hidden,
                 const float* __restrict__ W,
                 const float* __restrict__ bias,
                 float* __restrict__ out,
                 int Dd, int V, int LDO) {
    __shared__ __align__(16) unsigned short ldsA[BM * APITCH];
    __shared__ __align__(16) unsigned short ldsB[BN * APITCH];   // stored as B^T: [n][k]

    const int tid  = threadIdx.x;
    const int lane = tid & 31;
    const int wid  = tid >> 5;      // 8 waves
    const int wm   = wid >> 2;      // 0..1 -> 64-row strip
    const int wn   = wid & 3;       // 0..3 -> 32-col strip
    const int m0   = blockIdx.x * BM;
    const int v0   = blockIdx.y * BN;

    const int lrow  = lane & 15;
    const int khalf = (lane >> 4) * 8;   // bf16 A/B frag: lanes 16-31 start at K=8

    // Per-thread staging coordinates (fixed across k-steps)
    int arow[4], acol[4], bkr[4], bn4[4];
    #pragma unroll
    for (int i = 0; i < 4; ++i) {
        int f = tid + 256 * i;           // 1024 float4 per tile
        arow[i] = f >> 3;                // A: 8 float4 per 32-col row
        acol[i] = (f & 7) << 2;
        bkr[i]  = f >> 5;                // B: 32 float4 per 128-col row
        bn4[i]  = (f & 31) << 2;
    }

    v8f acc[4][2] = {};
    float4 ra[4], rb[4];

    auto load_tiles = [&](int k0) {
        #pragma unroll
        for (int i = 0; i < 4; ++i)
            ra[i] = *(const float4*)(hidden + (size_t)(m0 + arow[i]) * Dd + k0 + acol[i]);
        #pragma unroll
        for (int i = 0; i < 4; ++i) {
            int gc = v0 + bn4[i];
            const float* wr = W + (size_t)(k0 + bkr[i]) * V;
            if (gc + 3 < V) {
                rb[i] = *(const float4*)(wr + gc);
            } else {
                rb[i].x = (gc + 0 < V) ? wr[gc + 0] : 0.f;
                rb[i].y = (gc + 1 < V) ? wr[gc + 1] : 0.f;
                rb[i].z = (gc + 2 < V) ? wr[gc + 2] : 0.f;
                rb[i].w = (gc + 3 < V) ? wr[gc + 3] : 0.f;
            }
        }
    };

    load_tiles(0);

    for (int k0 = 0; k0 < Dd; k0 += BK) {
        // ---- commit staged registers to LDS (f32 -> bf16) ----
        #pragma unroll
        for (int i = 0; i < 4; ++i) {
            uint2 pk;
            pk.x = pack2_bf16(ra[i].x, ra[i].y);
            pk.y = pack2_bf16(ra[i].z, ra[i].w);
            *(uint2*)(&ldsA[arow[i] * APITCH + acol[i]]) = pk;
        }
        #pragma unroll
        for (int i = 0; i < 4; ++i) {
            ldsB[(bn4[i] + 0) * APITCH + bkr[i]] = bf16_hi(rb[i].x);
            ldsB[(bn4[i] + 1) * APITCH + bkr[i]] = bf16_hi(rb[i].y);
            ldsB[(bn4[i] + 2) * APITCH + bkr[i]] = bf16_hi(rb[i].z);
            ldsB[(bn4[i] + 3) * APITCH + bkr[i]] = bf16_hi(rb[i].w);
        }
        __syncthreads();

        // ---- prefetch next k-tile from global; latency hides under WMMAs ----
        if (k0 + BK < Dd) load_tiles(k0 + BK);

        // ---- fragments: two contiguous 8x bf16 chunks per lane (16B each) ----
        union Frag { v16bf v; uint4 q[2]; };
        Frag a[4], bf[2];
        #pragma unroll
        for (int mi = 0; mi < 4; ++mi) {
            const unsigned short* p = &ldsA[(wm * 64 + mi * 16 + lrow) * APITCH + khalf];
            a[mi].q[0] = *(const uint4*)(p);
            a[mi].q[1] = *(const uint4*)(p + 16);
        }
        #pragma unroll
        for (int ni = 0; ni < 2; ++ni) {
            const unsigned short* p = &ldsB[(wn * 32 + ni * 16 + lrow) * APITCH + khalf];
            bf[ni].q[0] = *(const uint4*)(p);
            bf[ni].q[1] = *(const uint4*)(p + 16);
        }
        #pragma unroll
        for (int mi = 0; mi < 4; ++mi)
            #pragma unroll
            for (int ni = 0; ni < 2; ++ni)
                acc[mi][ni] = __builtin_amdgcn_wmma_f32_16x16x32_bf16(
                    false, a[mi].v, false, bf[ni].v, (short)0, acc[mi][ni], false, false);
        __syncthreads();
    }

    // ---- epilogue: raw logits + bias ----
    #pragma unroll
    for (int mi = 0; mi < 4; ++mi) {
        #pragma unroll
        for (int ni = 0; ni < 2; ++ni) {
            int col = v0 + wn * 32 + ni * 16 + lrow;
            if (col < V) {
                float bv = bias[col];
                int rbase = m0 + wm * 64 + mi * 16 + ((lane >> 4) ? 8 : 0);
                #pragma unroll
                for (int r = 0; r < 8; ++r)
                    out[(size_t)(rbase + r) * LDO + col] = acc[mi][ni][r] + bv;
            }
        }
    }
}

// ---------------- copy gate: sigmoid(hidden @ w_copy + b_copy) ----------------
__global__ __launch_bounds__(256)
void copy_gate(const float* __restrict__ hidden, const float* __restrict__ w_copy,
               const float* __restrict__ b_copy, float* __restrict__ copyg, int Dd) {
    int n = blockIdx.x;
    float s = 0.f;
    for (int d = threadIdx.x; d < Dd; d += 256)
        s += hidden[(size_t)n * Dd + d] * w_copy[d];
    __shared__ float red[256];
    red[threadIdx.x] = s;
    __syncthreads();
    for (int off = 128; off > 0; off >>= 1) {
        if (threadIdx.x < off) red[threadIdx.x] += red[threadIdx.x + off];
        __syncthreads();
    }
    if (threadIdx.x == 0)
        copyg[n] = 1.0f / (1.0f + __expf(-(red[0] + b_copy[0])));
}

// ------------- per-row online softmax stats (skip PAD_IDX=1), fused scale -------------
__global__ __launch_bounds__(256)
void row_softmax_stats(const float* __restrict__ out, const float* __restrict__ copyg,
                       float* __restrict__ rowmax, float* __restrict__ scale,
                       int V, int LDO) {
    int n = blockIdx.x;
    const float* row = out + (size_t)n * LDO;
    float m = -INFINITY, s = 0.f;
    for (int v = threadIdx.x; v < V; v += 256) {
        if (v == 1) continue;                 // PAD masked to -inf -> prob 0
        float x = row[v];
        if (x > m) { s *= __expf(m - x); m = x; }
        s += __expf(x - m);
    }
    __shared__ float sm[256], ss[256];
    sm[threadIdx.x] = m; ss[threadIdx.x] = s;
    __syncthreads();
    for (int off = 128; off > 0; off >>= 1) {
        if (threadIdx.x < off) {
            float m1 = sm[threadIdx.x],       s1 = ss[threadIdx.x];
            float m2 = sm[threadIdx.x + off], s2 = ss[threadIdx.x + off];
            float mm = fmaxf(m1, m2);
            sm[threadIdx.x] = mm;
            ss[threadIdx.x] = s1 * __expf(m1 - mm) + s2 * __expf(m2 - mm);
        }
        __syncthreads();
    }
    if (threadIdx.x == 0) {
        rowmax[n] = sm[0];
        scale[n]  = (1.0f - copyg[n]) / ss[0];   // softmax norm * (1 - p_copy)
    }
}

// ---------------- finalize generate part: exp(x-max)*scale, PAD -> 0 ----------------
__global__ __launch_bounds__(256)
void finalize_gen(float* __restrict__ out, const float* __restrict__ rowmax,
                  const float* __restrict__ scale, int V, int LDO, int N) {
    size_t idx = (size_t)blockIdx.x * 256 + threadIdx.x;
    size_t total = (size_t)N * V;
    if (idx >= total) return;
    int n = (int)(idx / V);
    int v = (int)(idx % V);
    size_t o = (size_t)n * LDO + v;
    float x = out[o];
    out[o] = (v == 1) ? 0.0f : __expf(x - rowmax[n]) * scale[n];
}

// -------- copy distribution: out[n, V+cv] = copy[n] * sum_s attn[n,s]*src_map[s,b,cv] --------
__global__ __launch_bounds__(128)
void copy_dist(const float* __restrict__ attn, const float* __restrict__ src_map,
               const float* __restrict__ copyg, float* __restrict__ out,
               int S, int Bb, int CV, int V, int LDO) {
    int n  = blockIdx.x;
    int bb = n % Bb;                          // N is T-major: n = t*B + b
    __shared__ float at[512];
    for (int s = threadIdx.x; s < S; s += 128)
        at[s] = attn[(size_t)n * S + s];
    __syncthreads();
    float cg = copyg[n];
    for (int cv = threadIdx.x; cv < CV; cv += 128) {
        float acc = 0.f;
        for (int s = 0; s < S; ++s)
            acc += at[s] * src_map[((size_t)s * Bb + bb) * CV + cv];
        out[(size_t)n * LDO + V + cv] = cg * acc;
    }
}

extern "C" void kernel_launch(void* const* d_in, const int* in_sizes, int n_in,
                              void* d_out, int out_size, void* d_ws, size_t ws_size,
                              hipStream_t stream) {
    const float* hidden  = (const float*)d_in[0];
    const float* attn    = (const float*)d_in[1];
    const float* src_map = (const float*)d_in[2];
    const float* W       = (const float*)d_in[3];
    const float* bias    = (const float*)d_in[4];
    const float* w_copy  = (const float*)d_in[5];
    const float* b_copy  = (const float*)d_in[6];
    float* out = (float*)d_out;

    const int Dd  = in_sizes[5];              // w_copy is [D,1]
    const int V   = in_sizes[4];              // b is [V]
    const int N   = in_sizes[0] / Dd;         // hidden is [N,D]
    const int S   = in_sizes[1] / N;          // attn is [N,S]
    const int LDO = out_size / N;             // V + CV
    const int CV  = LDO - V;
    const int Bb  = in_sizes[2] / (S * CV);   // src_map is [S,B,CV]

    float* copyg  = (float*)d_ws;             // [N]
    float* rowmax = copyg + N;                // [N]
    float* scale  = rowmax + N;               // [N]

    copy_gate<<<N, 256, 0, stream>>>(hidden, w_copy, b_copy, copyg, Dd);

    dim3 grid((N + BM - 1) / BM, (V + BN - 1) / BN);
    gemm_logits<<<grid, 256, 0, stream>>>(hidden, W, bias, out, Dd, V, LDO);

    row_softmax_stats<<<N, 256, 0, stream>>>(out, copyg, rowmax, scale, V, LDO);

    size_t tot = (size_t)N * V;
    finalize_gen<<<(unsigned)((tot + 255) / 256), 256, 0, stream>>>(out, rowmax, scale, V, LDO, N);

    copy_dist<<<N, 128, 0, stream>>>(attn, src_map, copyg, out, S, Bb, CV, V, LDO);
}
